// Attention_1709396983811
// MI455X (gfx1250) — matrix operand order repaired and checked by reading, not compiled, not run on hardware
//
#include <hip/hip_runtime.h>

#define B_  2
#define S_  2048
#define E_  2048
#define H_  16
#define DH_ 128
#define M_  (B_*S_)

typedef __attribute__((ext_vector_type(16))) __bf16 v16bf;
typedef __attribute__((ext_vector_type(8)))  float  v8f;
typedef __attribute__((ext_vector_type(4)))  unsigned int u32x4;
typedef __attribute__((ext_vector_type(4)))  unsigned int vu4;
typedef __attribute__((ext_vector_type(8)))  int          vi8;
typedef __attribute__((ext_vector_type(4)))  int          vi4;

static __device__ __forceinline__ v8f wmma_bf16(v16bf a, v16bf b, v8f c) {
  return __builtin_amdgcn_wmma_f32_16x16x32_bf16(false, a, false, b, (short)0, c,
                                                 false, false);
}

// A fragment (16x32 bf16, MxK): lane row = lane&15; halves 0-7 at k=8*(lane>>4),
// halves 8-15 at +16.
static __device__ __forceinline__ v16bf frag_a_ld(const __bf16* rowp, int lhi) {
  union { v16bf v; u32x4 u[2]; } f;
  f.u[0] = *(const u32x4*)(rowp + 8*lhi);
  f.u[1] = *(const u32x4*)(rowp + 8*lhi + 16);
  return f.v;
}

// B fragment (32x16 bf16, KxN) staged N-major: lane col = lane&15; halves 0-7 at
// k=8*(lane>>4), halves 8-15 at +16.
static __device__ __forceinline__ v16bf frag_b_ld(const __bf16* rowp, int lhi) {
  union { v16bf v; u32x4 u[2]; } f;
  f.u[0] = *(const u32x4*)(rowp + 8*lhi);
  f.u[1] = *(const u32x4*)(rowp + 8*lhi + 16);
  return f.v;
}

// ---------------------------------------------------------------------------
// VALU-only 16-lane reductions via DPP (no LDS crossbar). Sequence:
// xor1 (quad_perm 1,0,3,2), xor2 (quad_perm 2,3,0,1), l^7 (row_half_mirror),
// l^15 (row_mirror). Stays within each 16-lane half of the wave.
// DPP ctrl must be an ICE -> template parameter.
// ---------------------------------------------------------------------------
template <int CTRL>
static __device__ __forceinline__ float dpp_mov_f32(float x) {
  int xi = __builtin_bit_cast(int, x);
  int r  = __builtin_amdgcn_update_dpp(0, xi, CTRL, 0xf, 0xf, true);
  return __builtin_bit_cast(float, r);
}
static __device__ __forceinline__ float red16_max(float x) {
  x = fmaxf(x, dpp_mov_f32<0xB1>(x));    // quad_perm [1,0,3,2]
  x = fmaxf(x, dpp_mov_f32<0x4E>(x));    // quad_perm [2,3,0,1]
  x = fmaxf(x, dpp_mov_f32<0x141>(x));   // row_half_mirror
  x = fmaxf(x, dpp_mov_f32<0x140>(x));   // row_mirror
  return x;
}
static __device__ __forceinline__ float red16_sum(float x) {
  x += dpp_mov_f32<0xB1>(x);
  x += dpp_mov_f32<0x4E>(x);
  x += dpp_mov_f32<0x141>(x);
  x += dpp_mov_f32<0x140>(x);
  return x;
}

// ---------------------------------------------------------------------------
// Tensor Data Mover: DMA a 2D bf16 tile (tile_d0 elems/row x tile_d1 rows,
// row stride t_stride0 elems in global) into LDS at byte offset lds_byte_off,
// inserting pad after every 2^(pad_int_code+1) DWORDs of (pad_amt_code+1) DWORDs.
// ---------------------------------------------------------------------------
static __device__ __forceinline__ void tdm_load_2d(
    unsigned lds_byte_off, const void* gptr,
    unsigned tile_d0, unsigned tile_d1,
    unsigned long long t_stride0,
    unsigned pad_int_code, unsigned pad_amt_code)
{
  unsigned long long ga = (unsigned long long)gptr;
  vu4 g0;
  g0[0] = 1u;                                                // count=1, user mode
  g0[1] = lds_byte_off;                                      // lds_addr
  g0[2] = (unsigned)(ga & 0xffffffffu);                      // global_addr lo
  g0[3] = (unsigned)((ga >> 32) & 0x1ffffffu) | (2u << 30);  // hi | type=2

  unsigned d[8] = {0,0,0,0,0,0,0,0};
  d[0] = (1u << 16)                  // data_size = 2 bytes
       | (1u << 20)                  // pad_enable
       | (pad_int_code << 22)
       | (pad_amt_code << 25);
  d[1] |= (tile_d0 & 0xffffu) << 16;         // tensor_dim0 lo (bits 63:48)
  d[2] |= (tile_d0 >> 16) & 0xffffu;         // tensor_dim0 hi (bits 79:64)
  d[2] |= (tile_d1 & 0xffffu) << 16;         // tensor_dim1 lo (bits 95:80)
  d[3] |= (tile_d1 >> 16) & 0xffffu;         // tensor_dim1 hi (bits 111:96)
  d[3] |= (tile_d0 & 0xffffu) << 16;         // tile_dim0 (bits 127:112)
  d[4] |= (tile_d1 & 0xffffu);               // tile_dim1 (bits 143:128)
  d[5]  = (unsigned)(t_stride0 & 0xffffffffu);     // tensor_dim0_stride lo
  d[6]  = (unsigned)((t_stride0 >> 32) & 0xffffu); // stride hi (48-bit)
  vi8 g1;
  #pragma unroll
  for (int i = 0; i < 8; ++i) g1[i] = (int)d[i];

  vi4 g2 = {0,0,0,0};
  vi4 g3 = {0,0,0,0};
#if defined(__clang_major__) && (__clang_major__ >= 23)
  vi8 g4 = {0,0,0,0,0,0,0,0};
  __builtin_amdgcn_tensor_load_to_lds(g0, g1, g2, g3, g4, 0);
#else
  __builtin_amdgcn_tensor_load_to_lds(g0, g1, g2, g3, 0);
#endif
}

// ---------------------------------------------------------------------------
// GEMM: out = A(f32, MxK) * W(f32, KxN) + bias, bf16 WMMA.
// BK=64, double-buffered dynamic LDS, 16 WMMA per barrier.
// MODE 0: f32 out[m*N+n]                                   (final projection)
// MODE 1: f32 headed out (bstride) + bf16 copy [b,h,s,dh]  (K)
// MODE 2: f32 headed out (bstride) + bf16 copy [b,h,dh,s]  (V, transposed)
// MODE 3: bf16-only copy [b,h,s,dh], scaled                (Q, 1/sqrt(Dh) folded)
// ---------------------------------------------------------------------------
#define BM 128
#define BN 128
#define BK 64
#define ASTR 72
#define BSTR 72
#define GEMM_SMEM_BYTES ((2*BM*ASTR + 2*BN*BSTR) * 2)

template <int MODE>
__global__ __launch_bounds__(256) void gemm_bias_bf16(
    const float* __restrict__ A, const float* __restrict__ W,
    const float* __restrict__ bias, float* __restrict__ out,
    __bf16* __restrict__ out2, int bstride, float scale)
{
  extern __shared__ __bf16 gsm[];
  __bf16* Asm0 = gsm;                    // 2 x [BM][ASTR]
  __bf16* Bsm0 = gsm + 2*BM*ASTR;        // 2 x [BN][BSTR]

  const int tid  = threadIdx.x;
  const int lane = tid & 31, w = tid >> 5;
  const int lhi  = lane >> 4, llo = lane & 15;
  const int wm   = (w >> 1) * 32;
  const int wn   = (w & 1) * 64;
  const int tm   = blockIdx.y * BM;
  const int tn   = blockIdx.x * BN;

  const v8f vzero = {0.f,0.f,0.f,0.f,0.f,0.f,0.f,0.f};
  v8f acc[2][4];
  #pragma unroll
  for (int i=0;i<2;++i)
    #pragma unroll
    for (int j=0;j<4;++j) acc[i][j] = vzero;

  const float* Ag = A + (size_t)tm * E_;
  const float* Wg = W + tn;

  auto stage = [&](int buf, int k0) {
    __bf16* As = Asm0 + buf*(BM*ASTR);
    __bf16* Bs = Bsm0 + buf*(BN*BSTR);
    #pragma unroll
    for (int v=0; v<8; ++v) {
      int c  = tid + v*256;
      int r  = c >> 4;                   // 16 float4 chunks per 64-elem row
      int cc = (c & 15) << 2;
      float4 d = *(const float4*)(Ag + (size_t)r*E_ + k0 + cc);
      __bf16* dst = &As[r*ASTR + cc];
      dst[0]=(__bf16)d.x; dst[1]=(__bf16)d.y; dst[2]=(__bf16)d.z; dst[3]=(__bf16)d.w;
    }
    #pragma unroll
    for (int v=0; v<8; ++v) {
      int c  = tid + v*256;
      int kr = c >> 5;                   // 32 float4 chunks per 128-elem n-row
      int nc = (c & 31) << 2;
      float4 d = *(const float4*)(Wg + (size_t)(k0+kr)*E_ + nc);
      Bs[(nc+0)*BSTR + kr] = (__bf16)d.x;
      Bs[(nc+1)*BSTR + kr] = (__bf16)d.y;
      Bs[(nc+2)*BSTR + kr] = (__bf16)d.z;
      Bs[(nc+3)*BSTR + kr] = (__bf16)d.w;
    }
  };

  stage(0, 0);
  __syncthreads();

  const int NK = E_ / BK;                // 32
  for (int kt = 0; kt < NK; ++kt) {
    const int cur = kt & 1;
    if (kt + 2 < NK) {                   // warm L2/WGP$ two steps ahead
      int kp = (kt + 2) * BK;
      #pragma unroll
      for (int v=0; v<8; v+=2) {
        int c = tid + v*256;
        __builtin_prefetch(Ag + (size_t)(c>>4)*E_ + kp + ((c&15)<<2), 0, 3);
        __builtin_prefetch(Wg + (size_t)(kp + (c>>5))*E_ + ((c&31)<<2), 0, 3);
      }
    }
    const __bf16* As = Asm0 + cur*(BM*ASTR);
    const __bf16* Bs = Bsm0 + cur*(BN*BSTR);
    #pragma unroll
    for (int ks2=0; ks2<2; ++ks2) {
      v16bf af[2], bfr[4];
      #pragma unroll
      for (int i=0;i<2;++i)
        af[i]  = frag_a_ld(&As[(wm + i*16 + llo)*ASTR + ks2*32], lhi);
      #pragma unroll
      for (int j=0;j<4;++j)
        bfr[j] = frag_b_ld(&Bs[(wn + j*16 + llo)*BSTR + ks2*32], lhi);
      #pragma unroll
      for (int i=0;i<2;++i)
        #pragma unroll
        for (int j=0;j<4;++j)
          acc[i][j] = wmma_bf16(af[i], bfr[j], acc[i][j]);
    }
    if (kt + 1 < NK) stage(cur ^ 1, (kt + 1) * BK);
    __syncthreads();
  }

  #pragma unroll
  for (int i=0;i<2;++i) {
    #pragma unroll
    for (int j=0;j<4;++j) {
      int ncol = tn + wn + j*16 + llo;
      float bv = bias[ncol];
      #pragma unroll
      for (int r=0;r<8;++r) {
        int m = tm + wm + i*16 + r + 8*lhi;
        float val = acc[i][j][r] + bv;
        if (MODE == 0) {
          out[(size_t)m * E_ + ncol] = val;
        } else {
          int b  = m / S_, s = m - b*S_;
          int hh = ncol >> 7;
          int dh = ncol & (DH_-1);
          if (MODE == 1 || MODE == 2)
            out[((size_t)(b*bstride + hh) * S_ + s) * DH_ + dh] = val;
          if (MODE == 1)
            out2[((size_t)(b*H_ + hh) * S_ + s) * DH_ + dh] = (__bf16)val;
          if (MODE == 2)
            out2[((size_t)(b*H_ + hh) * DH_ + dh) * S_ + s] = (__bf16)val;
          if (MODE == 3)
            out2[((size_t)(b*H_ + hh) * S_ + s) * DH_ + dh] = (__bf16)(val * scale);
        }
      }
    }
  }
}

// ---------------------------------------------------------------------------
// Flash attention, bf16 Q/K/Vt inputs, TDM-staged + double-buffered K/V tiles.
// One block = (b, h, 128 q rows); 8 waves x 16 q-rows each.
// ---------------------------------------------------------------------------
#define QSTR 136   // 128 halves + 16B TDM pad -> 272B rows (codes 5/3)
#define VSTR 72    // 64 halves + 16B TDM pad  -> 144B rows (codes 4/3)
#define ATT_SMEM_HALVES (128*QSTR + 2*64*QSTR + 2*128*VSTR + 8*16*VSTR)

__global__ __launch_bounds__(256) void flash_attn(
    const __bf16* __restrict__ Qb, const __bf16* __restrict__ Kb,
    const __bf16* __restrict__ Vtb, float* __restrict__ Y)
{
  extern __shared__ __bf16 smem[];
  __bf16* Qs     = smem;                                   // [128][QSTR]
  __bf16* KsBuf  = Qs + 128*QSTR;                          // 2 x [64][QSTR]
  __bf16* VtBuf  = KsBuf + 2*64*QSTR;                      // 2 x [128][VSTR]
  __bf16* Ps     = VtBuf + 2*128*VSTR;                     // [8][16][VSTR]

  const unsigned lds0    = __builtin_amdgcn_groupstaticsize();
  const unsigned qs_off  = lds0;
  const unsigned ks_off0 = qs_off + 128*QSTR*2;
  const unsigned vs_off0 = ks_off0 + 2*64*QSTR*2;

  const int bid = blockIdx.x;
  const int qt  = bid & 15;
  const int hh  = (bid >> 4) & (H_-1);
  const int b   = bid >> 8;

  const __bf16* Qg = Qb  + ((size_t)(b*H_ + hh) * S_ + qt*128) * DH_;
  const __bf16* Kg = Kb  + ((size_t)(b*H_ + hh) * S_) * DH_;
  const __bf16* Vg = Vtb + ((size_t)(b*H_ + hh) * DH_) * S_;
  float*        Yg = Y   + ((size_t)(b*S_) + qt*128) * E_ + hh*DH_;

  const int tid  = threadIdx.x;
  const int lane = tid & 31, w = tid >> 5;
  const int lhi  = lane >> 4, llo = lane & 15;
  const int qrow = w * 16;
  const int jend = (qt + 1) * 2;

  // prologue: Q tile + first K/V tiles via TDM (issued by wave 0 only)
  if (w == 0) {
    tdm_load_2d(qs_off,  Qg, 128, 128, 128, 5, 3);
    tdm_load_2d(ks_off0, Kg, 128,  64, 128, 5, 3);
    tdm_load_2d(vs_off0, Vg,  64, 128, S_,  4, 3);
  }

  float rowmax[8], rowsum[8], fscale[8];
  const v8f vzero = {0.f,0.f,0.f,0.f,0.f,0.f,0.f,0.f};
  v8f accO[8];
  #pragma unroll
  for (int r=0;r<8;++r){ rowmax[r] = -3.0e38f; rowsum[r] = 0.f; accO[r] = vzero; }

  __bf16* Pw = Ps + w*16*VSTR;
  v16bf qf[4];                            // Q A-fragments, resident after j==0

  for (int j = 0; j < jend; ++j) {
    const int cur = j & 1;
    __bf16* Ks  = KsBuf + cur*64*QSTR;
    __bf16* Vts = VtBuf + cur*128*VSTR;

    if (w == 0) {
      if (j + 1 < jend) {                 // issue next tiles, wait for current
        int nxt = cur ^ 1;
        tdm_load_2d(ks_off0 + (unsigned)nxt*64*QSTR*2,
                    Kg + (size_t)((j+1)*64)*DH_, 128, 64, 128, 5, 3);
        tdm_load_2d(vs_off0 + (unsigned)nxt*128*VSTR*2,
                    Vg + (j+1)*64, 64, 128, S_, 4, 3);
        __builtin_amdgcn_s_wait_tensorcnt(2);
      } else {
        __builtin_amdgcn_s_wait_tensorcnt(0);
      }
    }
    __syncthreads();                      // current tiles visible to all waves

    if (j == 0) {                         // hoist Q fragments into registers
      #pragma unroll
      for (int ks=0; ks<4; ++ks)
        qf[ks] = frag_a_ld(&Qs[(qrow + llo)*QSTR + ks*32], lhi);
    }

    // scores: 16 q-rows x 64 keys (K-dim = DH = 128, 4 steps of 32)
    v8f sacc[4];
    #pragma unroll
    for (int nt=0;nt<4;++nt) sacc[nt] = vzero;
    #pragma unroll
    for (int ks=0; ks<4; ++ks) {
      #pragma unroll
      for (int nt=0; nt<4; ++nt) {
        v16bf bk = frag_b_ld(&Ks[(nt*16 + llo)*QSTR + ks*32], lhi);
        sacc[nt] = wmma_bf16(qf[ks], bk, sacc[nt]);
      }
    }

    if (j >= qt*2) {                      // causal mask on diagonal tiles
      #pragma unroll
      for (int nt=0;nt<4;++nt)
        #pragma unroll
        for (int r=0;r<8;++r) {
          int qg = qt*128 + qrow + r + 8*lhi;
          int kg = j*64 + nt*16 + llo;
          if (kg > qg) sacc[nt][r] = -3.0e38f;
        }
    }

    // online softmax (row r lives in vgpr r across 16 lanes of this half)
    #pragma unroll
    for (int r=0;r<8;++r) {
      float mloc = fmaxf(fmaxf(sacc[0][r], sacc[1][r]),
                         fmaxf(sacc[2][r], sacc[3][r]));
      mloc = red16_max(mloc);
      float mnew = fmaxf(rowmax[r], mloc);
      fscale[r] = __expf(rowmax[r] - mnew);
      rowmax[r] = mnew;
      float psum = 0.f;
      #pragma unroll
      for (int nt=0;nt<4;++nt) {
        float p = __expf(sacc[nt][r] - mnew);
        sacc[nt][r] = p;
        psum += p;
      }
      psum = red16_sum(psum);
      rowsum[r] = rowsum[r]*fscale[r] + psum;
    }

    #pragma unroll
    for (int nt=0;nt<8;++nt)
      #pragma unroll
      for (int r=0;r<8;++r) accO[nt][r] *= fscale[r];

    // P -> per-wave LDS strip (C layout -> A layout), same-wave DS in-order
    #pragma unroll
    for (int nt=0;nt<4;++nt)
      #pragma unroll
      for (int r=0;r<8;++r)
        Pw[(r + 8*lhi)*VSTR + nt*16 + llo] = (__bf16)sacc[nt][r];

    // PV: 16x128 += (16x64) * (64x128)
    #pragma unroll
    for (int kk=0; kk<2; ++kk) {
      v16bf ap = frag_a_ld(&Pw[llo*VSTR + kk*32], lhi);
      #pragma unroll
      for (int nt=0; nt<8; ++nt) {
        v16bf bv = frag_b_ld(&Vts[(nt*16 + llo)*VSTR + kk*32], lhi);
        accO[nt] = wmma_bf16(ap, bv, accO[nt]);
      }
    }
    __syncthreads();                      // all reads done before overwrite
  }

  #pragma unroll
  for (int r=0;r<8;++r) {
    float inv = 1.0f / rowsum[r];
    int m = qrow + r + 8*lhi;
    #pragma unroll
    for (int nt=0;nt<8;++nt) {
      int dh = nt*16 + llo;
      Yg[(size_t)m*E_ + dh] = accO[nt][r] * inv;
    }
  }
}

// ---------------------------------------------------------------------------
extern "C" void kernel_launch(void* const* d_in, const int* in_sizes, int n_in,
                              void* d_out, int out_size, void* d_ws, size_t ws_size,
                              hipStream_t stream) {
  (void)in_sizes; (void)n_in; (void)out_size; (void)ws_size;

  const float* x  = (const float*)d_in[0];
  const float* Wq = (const float*)d_in[1];
  const float* bq = (const float*)d_in[2];
  const float* Wk = (const float*)d_in[3];
  const float* bk = (const float*)d_in[4];
  const float* Wv = (const float*)d_in[5];
  const float* bv = (const float*)d_in[6];
  const float* Wo = (const float*)d_in[7];
  const float* bo = (const float*)d_in[8];

  float* y  = (float*)d_out;                        // [B,S,E] f32
  float* kv = y + (size_t)B_ * S_ * E_;             // [B,2,H,S,DH] f32

  const size_t NQ = (size_t)B_ * H_ * S_ * DH_;     // 8,388,608 elements
  __bf16* Qb  = (__bf16*)d_ws;                      // [B,H,S,DH] bf16, pre-scaled
  __bf16* Kb  = Qb + NQ;                            // [B,H,S,DH] bf16
  __bf16* Vtb = Kb + NQ;                            // [B,H,DH,S] bf16 (transposed)
  float*  Yws = (float*)(Vtb + NQ);                 // [B,S,E] f32

  const float qscale = 0.08838834764831845f;        // 1/sqrt(128)

  dim3 grid(E_/BN, M_/BM);
  gemm_bias_bf16<3><<<grid, 256, GEMM_SMEM_BYTES, stream>>>(x, Wq, bq, nullptr,
                                                            Qb, 0, qscale);
  gemm_bias_bf16<1><<<grid, 256, GEMM_SMEM_BYTES, stream>>>(x, Wk, bk, kv,
                                                            Kb, 2*H_, 1.f);
  gemm_bias_bf16<2><<<grid, 256, GEMM_SMEM_BYTES, stream>>>(x, Wv, bv,
                                                            kv + (size_t)H_*S_*DH_,
                                                            Vtb, 2*H_, 1.f);

  size_t smem_bytes = (size_t)ATT_SMEM_HALVES * 2;  // 124,928 B dynamic LDS
  flash_attn<<<B_*H_*(S_/128), 256, smem_bytes, stream>>>(Qb, Kb, Vtb, Yws);

  gemm_bias_bf16<0><<<grid, 256, GEMM_SMEM_BYTES, stream>>>(Yws, Wo, bo, y,
                                                            nullptr, 0, 1.f);
}
